// MultiHeadedAttention_36378372997688
// MI455X (gfx1250) — compile-verified, matrix operand order checked by reading, not run on hardware
//
#include <hip/hip_runtime.h>

#define EMB   1024
#define HEADS 16
#define HDIM  64
#define SEQ   2048
#define BATCH 4
#define ROWS  (BATCH * SEQ)   // 8192

typedef __attribute__((ext_vector_type(16))) _Float16 v16h;
typedef __attribute__((ext_vector_type(8)))  float    v8f;

union H8 { float4 f4; _Float16 h[8]; };

__device__ __forceinline__ v8f wmma16(v16h a, v16h b, v8f c) {
    return __builtin_amdgcn_wmma_f32_16x16x32_f16(false, a, false, b, (short)0, c,
                                                  false, false);
}

// 16-bit 16x32 operand fragment (A and B share the per-lane pattern, per the
// ISA A-matrix table and the 4-VGPR granularity of the TR16 loads):
// lane holds row/col (lane%16); halves are K[ko..ko+7] ++ K[16+ko..16+ko+7],
// ko = (lane>=16) ? 8 : 0.  `p` points at 32 contiguous f16 K-values.
__device__ __forceinline__ v16h load_frag(const _Float16* p, int lane) {
    const int ko = (lane & 16) ? 8 : 0;
    v16h f;
#pragma unroll
    for (int i = 0; i < 8; ++i) {
        f[i]     = p[ko + i];
        f[8 + i] = p[16 + ko + i];
    }
    return f;
}

// ---------------------------------------------------------------------------
// Kernel 1: Y = X @ W + bias (f32 in, f16 out in [b,h,s,d] layout, scaled).
// Block 256 (8 waves, 4Mx2N); block tile 128x128, BK=32, double-buffered LDS,
// global loads for tile i+1 issued before the WMMA block of tile i.
// LDS fills are packed: 8 halves per ds_store_b128.
// ---------------------------------------------------------------------------
__global__ __launch_bounds__(256)
void qkv_proj_kernel(const float* __restrict__ X, const float* __restrict__ W,
                     const float* __restrict__ bias, _Float16* __restrict__ out,
                     float scale) {
    __shared__ __align__(16) _Float16 ax[2][128 * 40];  // X tile [m][k]
    __shared__ __align__(16) _Float16 wt[2][128 * 40];  // W tile transposed [n][k]

    const int t    = threadIdx.x;
    const int lane = t & 31;
    const int wid  = t >> 5;
    const int wm   = wid >> 1, wn = wid & 1;
    const int n0   = blockIdx.x * 128;
    const int m0   = blockIdx.y * 128;
    const int lq   = lane & 15;
    const int wn8  = t & 127;            // W-tile: column n handled by thread

    v8f acc[2][4] = {};

    // Prologue: fill buffer 0 with tile kb=0 (packed b128 stores).
#pragma unroll
    for (int c = 0; c < 2; ++c) {
        int idx = c * 256 + t, r = idx >> 2, c8 = (idx & 3) * 8;
        H8 u;
#pragma unroll
        for (int i = 0; i < 8; ++i)
            u.h[i] = (_Float16)X[(size_t)(m0 + r) * EMB + c8 + i];
        *reinterpret_cast<float4*>(&ax[0][r * 40 + c8]) = u.f4;
    }
#pragma unroll
    for (int c = 0; c < 2; ++c) {
        int k0 = ((t >> 7) + c * 2) * 8;
        H8 u;
#pragma unroll
        for (int i = 0; i < 8; ++i)
            u.h[i] = (_Float16)W[(size_t)(k0 + i) * EMB + n0 + wn8];
        *reinterpret_cast<float4*>(&wt[0][wn8 * 40 + k0]) = u.f4;
    }

    for (int kb = 0, buf = 0; kb < EMB; kb += 32, buf ^= 1) {
        __syncthreads();  // buf filled everywhere; previous compute finished

        float xr[16], wr[16];
        const bool more = (kb + 32) < EMB;
        if (more) {  // issue next tile's global loads (overlap with WMMA below)
#pragma unroll
            for (int c = 0; c < 2; ++c) {
                int idx = c * 256 + t, r = idx >> 2, c8 = (idx & 3) * 8;
#pragma unroll
                for (int i = 0; i < 8; ++i)
                    xr[c * 8 + i] = X[(size_t)(m0 + r) * EMB + kb + 32 + c8 + i];
            }
#pragma unroll
            for (int c = 0; c < 2; ++c) {
                int k0 = ((t >> 7) + c * 2) * 8;
#pragma unroll
                for (int i = 0; i < 8; ++i)
                    wr[c * 8 + i] = W[(size_t)(kb + 32 + k0 + i) * EMB + n0 + wn8];
            }
        }

        v16h afr[2], bfr[4];
#pragma unroll
        for (int tm = 0; tm < 2; ++tm)
            afr[tm] = load_frag(&ax[buf][(wm * 32 + tm * 16 + lq) * 40], lane);
#pragma unroll
        for (int tn = 0; tn < 4; ++tn)
            bfr[tn] = load_frag(&wt[buf][(wn * 64 + tn * 16 + lq) * 40], lane);
#pragma unroll
        for (int tm = 0; tm < 2; ++tm)
#pragma unroll
            for (int tn = 0; tn < 4; ++tn)
                acc[tm][tn] = wmma16(afr[tm], bfr[tn], acc[tm][tn]);

        if (more) {  // pack + store next tile into the other buffer
#pragma unroll
            for (int c = 0; c < 2; ++c) {
                int idx = c * 256 + t, r = idx >> 2, c8 = (idx & 3) * 8;
                H8 u;
#pragma unroll
                for (int i = 0; i < 8; ++i) u.h[i] = (_Float16)xr[c * 8 + i];
                *reinterpret_cast<float4*>(&ax[buf ^ 1][r * 40 + c8]) = u.f4;
            }
#pragma unroll
            for (int c = 0; c < 2; ++c) {
                int k0 = ((t >> 7) + c * 2) * 8;
                H8 u;
#pragma unroll
                for (int i = 0; i < 8; ++i) u.h[i] = (_Float16)wr[c * 8 + i];
                *reinterpret_cast<float4*>(&wt[buf ^ 1][wn8 * 40 + k0]) = u.f4;
            }
        }
    }

#pragma unroll
    for (int tm = 0; tm < 2; ++tm)
#pragma unroll
        for (int tn = 0; tn < 4; ++tn)
#pragma unroll
            for (int v = 0; v < 8; ++v) {
                int m = m0 + wm * 32 + tm * 16 + v + ((lane >> 4) << 3);
                int n = n0 + wn * 64 + tn * 16 + lq;
                float y = (acc[tm][tn][v] + bias[n]) * scale;
                int b = m >> 11, s = m & (SEQ - 1);
                int h = n >> 6, d = n & 63;
                out[((size_t)(b * HEADS + h) * SEQ + s) * HDIM + d] = (_Float16)y;
            }
}

// ---------------------------------------------------------------------------
// Kernel 2: flash attention, exp2-domain softmax (scale*log2e folded into Q).
// 64-key tiles of K and V staged via global_load_async_to_lds_b128 into
// double-buffered LDS (ASYNCcnt), prefetching tile i+1 during compute of i.
// grid (SEQ/128, BATCH*HEADS), block 256; each wave owns 16 query rows.
// ---------------------------------------------------------------------------
__global__ __launch_bounds__(256)
void attn_kernel(const _Float16* __restrict__ Q, const _Float16* __restrict__ K,
                 const _Float16* __restrict__ V, _Float16* __restrict__ AV) {
    __shared__ __align__(16) _Float16 kraw[2][64 * 72];  // K tile [key][d]
    __shared__ __align__(16) _Float16 vraw[2][64 * 72];  // V tile [key][d]
    __shared__ __align__(16) _Float16 vt[64 * 72];       // V transposed [d][key]
    __shared__ __align__(16) _Float16 psc[8 * 16 * 64];  // per-wave P scratch

    const int t    = threadIdx.x;
    const int lane = t & 31;
    const int wid  = t >> 5;
    const int bh   = blockIdx.y;
    const int b    = bh >> 4, h = bh & 15;
    const int q0   = blockIdx.x * 128 + wid * 16;
    const int lq   = lane & 15;

    const _Float16* qg = Q + (size_t)bh * SEQ * HDIM;
    const _Float16* kg = K + (size_t)bh * SEQ * HDIM;
    const _Float16* vg = V + (size_t)bh * SEQ * HDIM;

    // Q fragments stay in registers for the whole kernel.
    const _Float16* qrow = qg + (size_t)(q0 + lq) * HDIM;
    v16h qf0 = load_frag(qrow, lane);
    v16h qf1 = load_frag(qrow + 32, lane);

    v8f o[4] = {};
    float mrow[8], lrow[8];
#pragma unroll
    for (int v = 0; v < 8; ++v) { mrow[v] = -1e30f; lrow[v] = 0.f; }

    _Float16* pw = &psc[wid * 16 * 64];

    // Async-stage one 64x64 f16 K tile + V tile: 512 16B chunks each,
    // 2 chunks per thread per tile -> 4 async instructions per wave.
    auto issue_tile = [&](int kt0, int bufi) {
#pragma unroll
        for (int c = 0; c < 2; ++c) {
            int idx = c * 256 + t;             // 16B chunk id, 0..511
            int key = idx >> 3, d0 = (idx & 7) * 8;
            const _Float16* gK = kg + (size_t)(kt0 + key) * HDIM + d0;
            const _Float16* gV = vg + (size_t)(kt0 + key) * HDIM + d0;
            unsigned lK = (unsigned)(size_t)&kraw[bufi][key * 72 + d0];
            unsigned lV = (unsigned)(size_t)&vraw[bufi][key * 72 + d0];
            asm volatile("global_load_async_to_lds_b128 %0, %1, off"
                         :: "v"(lK), "v"(gK) : "memory");
            asm volatile("global_load_async_to_lds_b128 %0, %1, off"
                         :: "v"(lV), "v"(gV) : "memory");
        }
    };

    issue_tile(0, 0);

    for (int kt0 = 0, buf = 0; kt0 < SEQ; kt0 += 64, buf ^= 1) {
        asm volatile("s_wait_asynccnt 0x0" ::: "memory");
        __syncthreads();  // current buffers resident; prior compute finished

        if (kt0 + 64 < SEQ) issue_tile(kt0 + 64, buf ^ 1);  // prefetch next

        // Transpose vraw[buf] -> vt (amortized across all 8 waves).
#pragma unroll
        for (int c = 0; c < 2; ++c) {
            int idx = c * 256 + t;
            int key = idx >> 3, d0 = (idx & 7) * 8;
            H8 u;
            u.f4 = *reinterpret_cast<const float4*>(&vraw[buf][key * 72 + d0]);
#pragma unroll
            for (int i = 0; i < 8; ++i) vt[(d0 + i) * 72 + key] = u.h[i];
        }
        __syncthreads();  // vt ready

        // S = Q K^T: four 16x16 key sub-tiles, 2 K-steps each over d=64.
        v8f s[4];
#pragma unroll
        for (int st = 0; st < 4; ++st) {
            v8f a = {};
            v16h b0 = load_frag(&kraw[buf][(st * 16 + lq) * 72 + 0], lane);
            v16h b1 = load_frag(&kraw[buf][(st * 16 + lq) * 72 + 32], lane);
            a = wmma16(qf0, b0, a);
            a = wmma16(qf1, b1, a);
            s[st] = a;
        }

        // Online softmax (exp2 domain); row r = v + 8*(lane>=16).
#pragma unroll
        for (int v = 0; v < 8; ++v) {
            float mx = fmaxf(fmaxf(s[0][v], s[1][v]), fmaxf(s[2][v], s[3][v]));
#pragma unroll
            for (int w = 1; w < 16; w <<= 1)
                mx = fmaxf(mx, __shfl_xor(mx, w, 32));
            float mnew = fmaxf(mrow[v], mx);
            float corr = __builtin_amdgcn_exp2f(mrow[v] - mnew);
            float ps = 0.f;
#pragma unroll
            for (int st = 0; st < 4; ++st) {
                float p = __builtin_amdgcn_exp2f(s[st][v] - mnew);
                s[st][v] = p;
                ps += p;
            }
#pragma unroll
            for (int w = 1; w < 16; w <<= 1)
                ps += __shfl_xor(ps, w, 32);
            lrow[v] = lrow[v] * corr + ps;
            mrow[v] = mnew;
#pragma unroll
            for (int tt = 0; tt < 4; ++tt) o[tt][v] *= corr;
        }

        // P tile: C-layout -> A-layout via wave-private LDS round trip.
#pragma unroll
        for (int v = 0; v < 8; ++v) {
            int r = v + ((lane >> 4) << 3);
#pragma unroll
            for (int st = 0; st < 4; ++st)
                pw[r * 64 + st * 16 + lq] = (_Float16)s[st][v];
        }
        asm volatile("s_wait_dscnt 0x0" ::: "memory");

        v16h pf0 = load_frag(&pw[lq * 64 + 0], lane);
        v16h pf1 = load_frag(&pw[lq * 64 + 32], lane);
#pragma unroll
        for (int tt = 0; tt < 4; ++tt) {
            v16h bv0 = load_frag(&vt[(tt * 16 + lq) * 72 + 0], lane);
            v16h bv1 = load_frag(&vt[(tt * 16 + lq) * 72 + 32], lane);
            o[tt] = wmma16(pf0, bv0, o[tt]);
            o[tt] = wmma16(pf1, bv1, o[tt]);
        }
        // no trailing barrier: next iteration's top barrier orders vt reuse
    }

    // Normalize and store f16 in [b, s, h*64+d] layout.
#pragma unroll
    for (int v = 0; v < 8; ++v) {
        float inv = 1.0f / lrow[v];
        int q = q0 + v + ((lane >> 4) << 3);
        size_t rowbase = ((size_t)b * SEQ + q) * EMB + h * HDIM;
#pragma unroll
        for (int tt = 0; tt < 4; ++tt)
            AV[rowbase + tt * 16 + lq] = (_Float16)(o[tt][v] * inv);
    }
}

// ---------------------------------------------------------------------------
// Kernel 3: out = AV @ Wo + bo, f32 output.  Same pipelined tiling as K1.
// ---------------------------------------------------------------------------
__global__ __launch_bounds__(256)
void out_proj_kernel(const _Float16* __restrict__ AV, const float* __restrict__ W,
                     const float* __restrict__ bias, float* __restrict__ out) {
    __shared__ __align__(16) _Float16 ax[2][128 * 40];
    __shared__ __align__(16) _Float16 wt[2][128 * 40];

    const int t    = threadIdx.x;
    const int lane = t & 31;
    const int wid  = t >> 5;
    const int wm   = wid >> 1, wn = wid & 1;
    const int n0   = blockIdx.x * 128;
    const int m0   = blockIdx.y * 128;
    const int lq   = lane & 15;
    const int r    = t >> 1, cc = (t & 1) * 16;  // AV staging map (16 halves)
    const int wn8  = t & 127;

    v8f acc[2][4] = {};

    {   // prologue
        const float4* g = reinterpret_cast<const float4*>(
            AV + (size_t)(m0 + r) * EMB + cc);
        float4* l = reinterpret_cast<float4*>(&ax[0][r * 40 + cc]);
        l[0] = g[0];
        l[1] = g[1];
#pragma unroll
        for (int c = 0; c < 2; ++c) {
            int k0 = ((t >> 7) + c * 2) * 8;
            H8 u;
#pragma unroll
            for (int i = 0; i < 8; ++i)
                u.h[i] = (_Float16)W[(size_t)(k0 + i) * EMB + n0 + wn8];
            *reinterpret_cast<float4*>(&wt[0][wn8 * 40 + k0]) = u.f4;
        }
    }

    for (int kb = 0, buf = 0; kb < EMB; kb += 32, buf ^= 1) {
        __syncthreads();

        float4 xr[2];
        float wr[16];
        const bool more = (kb + 32) < EMB;
        if (more) {
            const float4* g = reinterpret_cast<const float4*>(
                AV + (size_t)(m0 + r) * EMB + kb + 32 + cc);
            xr[0] = g[0];
            xr[1] = g[1];
#pragma unroll
            for (int c = 0; c < 2; ++c) {
                int k0 = ((t >> 7) + c * 2) * 8;
#pragma unroll
                for (int i = 0; i < 8; ++i)
                    wr[c * 8 + i] = W[(size_t)(kb + 32 + k0 + i) * EMB + n0 + wn8];
            }
        }

        v16h afr[2], bfr[4];
#pragma unroll
        for (int tm = 0; tm < 2; ++tm)
            afr[tm] = load_frag(&ax[buf][(wm * 32 + tm * 16 + lq) * 40], lane);
#pragma unroll
        for (int tn = 0; tn < 4; ++tn)
            bfr[tn] = load_frag(&wt[buf][(wn * 64 + tn * 16 + lq) * 40], lane);
#pragma unroll
        for (int tm = 0; tm < 2; ++tm)
#pragma unroll
            for (int tn = 0; tn < 4; ++tn)
                acc[tm][tn] = wmma16(afr[tm], bfr[tn], acc[tm][tn]);

        if (more) {
            float4* l = reinterpret_cast<float4*>(&ax[buf ^ 1][r * 40 + cc]);
            l[0] = xr[0];
            l[1] = xr[1];
#pragma unroll
            for (int c = 0; c < 2; ++c) {
                int k0 = ((t >> 7) + c * 2) * 8;
                H8 u;
#pragma unroll
                for (int i = 0; i < 8; ++i) u.h[i] = (_Float16)wr[c * 8 + i];
                *reinterpret_cast<float4*>(&wt[buf ^ 1][wn8 * 40 + k0]) = u.f4;
            }
        }
    }

#pragma unroll
    for (int tm = 0; tm < 2; ++tm)
#pragma unroll
        for (int tn = 0; tn < 4; ++tn)
#pragma unroll
            for (int v = 0; v < 8; ++v) {
                int m = m0 + wm * 32 + tm * 16 + v + ((lane >> 4) << 3);
                int n = n0 + wn * 64 + tn * 16 + lq;
                out[(size_t)m * EMB + n] = acc[tm][tn][v] + bias[n];
            }
}

extern "C" void kernel_launch(void* const* d_in, const int* in_sizes, int n_in,
                              void* d_out, int out_size, void* d_ws, size_t ws_size,
                              hipStream_t stream) {
    const float* X  = (const float*)d_in[0];
    const float* Wq = (const float*)d_in[1];
    const float* bq = (const float*)d_in[2];
    const float* Wk = (const float*)d_in[3];
    const float* bk = (const float*)d_in[4];
    const float* Wv = (const float*)d_in[5];
    const float* bv = (const float*)d_in[6];
    const float* Wo = (const float*)d_in[7];
    const float* bo = (const float*)d_in[8];

    _Float16* ws  = (_Float16*)d_ws;
    const size_t PLANE = (size_t)ROWS * EMB;  // 8M f16 = 16 MB
    _Float16* Qh  = ws;
    _Float16* Kh  = ws + PLANE;
    _Float16* Vh  = ws + 2 * PLANE;
    _Float16* AVh = ws + 3 * PLANE;

    // Fold softmax scale AND log2(e) into Q so exp() becomes exp2().
    const float qscale = 0.125f * 1.4426950408889634f;

    dim3 gproj(EMB / 128, ROWS / 128);
    qkv_proj_kernel<<<gproj, 256, 0, stream>>>(X, Wq, bq, Qh, qscale);
    qkv_proj_kernel<<<gproj, 256, 0, stream>>>(X, Wk, bk, Kh, 1.0f);
    qkv_proj_kernel<<<gproj, 256, 0, stream>>>(X, Wv, bv, Vh, 1.0f);

    attn_kernel<<<dim3(SEQ / 128, BATCH * HEADS), 256, 0, stream>>>(Qh, Kh, Vh, AVh);

    out_proj_kernel<<<gproj, 256, 0, stream>>>(AVh, Wo, bo, (float*)d_out);
}
